// Decoder_16784732193242
// MI455X (gfx1250) — compile-verified, matrix operand order checked by reading, not run on hardware
//
#include <hip/hip_runtime.h>
#include <hip/hip_bf16.h>
#include <math.h>

// ---- problem constants -----------------------------------------------------
#define Bc 4
#define Tc 500
#define Uc 100
#define Dc 512
#define Hc 512
#define Lc 2
#define Vc 64
#define TU (Tc*Uc)            // 50000 rows per batch (multiple of 16 -> tiles never cross b)
#define MTOT (Bc*TU)          // 200000 rows total
#define MTILES (MTOT/16)      // 12500 wmma M-tiles
#define OUT0 ((size_t)Bc*Tc*Uc*Vc)   // 12,800,000 floats, then h(4096) then c(4096)

typedef __attribute__((ext_vector_type(16))) _Float16 v16h;
typedef __attribute__((ext_vector_type(8)))  _Float16 v8h;
typedef __attribute__((ext_vector_type(8)))  float    v8f;

__device__ __forceinline__ float sigf(float x) { return 1.0f / (1.0f + expf(-x)); }

// Fast tanh: prefer the gfx1250 V_TANH_F32 TRANS op (co-executes with WMMA);
// fall back to a v_exp_f32-based identity (4 ops, ~1e-7 rel err, clamp keeps
// exp finite; joint inputs are |x| <~ 5 so the clamp never changes results).
__device__ __forceinline__ float fast_tanh(float x) {
#if __has_builtin(__builtin_amdgcn_tanhf)
  return __builtin_amdgcn_tanhf(x);
#elif __has_builtin(__builtin_amdgcn_tanh_f32)
  return __builtin_amdgcn_tanh_f32(x);
#else
  float xc = fminf(10.0f, fmaxf(-10.0f, x));
  float e  = __expf(2.0f * xc);
#if __has_builtin(__builtin_amdgcn_rcpf)
  return (e - 1.0f) * __builtin_amdgcn_rcpf(e + 1.0f);
#else
  return (e - 1.0f) / (e + 1.0f);
#endif
#endif
}

// Load 16 f16 A/B-fragment elements from a row-major f32 row.
// k pattern per ISA 16-bit 16x32 layout: [kb..kb+7] and [kb+16..kb+23],
// kb = (lane>>4)*8, row base already offset by k0 (multiple of 32).
__device__ __forceinline__ v16h load_frag_f32(const float* p, int kb) {
  const float4* q = (const float4*)p;
  int i0 = kb >> 2;                       // kb in {0,8} -> i0 in {0,2}
  float4 a0 = q[i0],     a1 = q[i0 + 1];  // k = kb..kb+7
  float4 a2 = q[i0 + 4], a3 = q[i0 + 5];  // k = kb+16..kb+23
  v16h r;
  r[0]=(_Float16)a0.x; r[1]=(_Float16)a0.y; r[2]=(_Float16)a0.z; r[3]=(_Float16)a0.w;
  r[4]=(_Float16)a1.x; r[5]=(_Float16)a1.y; r[6]=(_Float16)a1.z; r[7]=(_Float16)a1.w;
  r[8]=(_Float16)a2.x; r[9]=(_Float16)a2.y; r[10]=(_Float16)a2.z; r[11]=(_Float16)a2.w;
  r[12]=(_Float16)a3.x; r[13]=(_Float16)a3.y; r[14]=(_Float16)a3.z; r[15]=(_Float16)a3.w;
  return r;
}

// ---- kernel 1: embedding + 2-layer LSTM (persistent, one block per batch) --
__global__ __launch_bounds__(1024)
void k_lstm(const int* __restrict__ ids, const float* __restrict__ emb,
            const float* __restrict__ w_ih, const float* __restrict__ w_hh,
            const float* __restrict__ b_ih, const float* __restrict__ b_hh,
            float* __restrict__ ys, float* __restrict__ hc_out) {
  __shared__ float xv[Hc], hv[Hc], cv[Hc], gates[4*Hc];
  const int b = blockIdx.x, tid = threadIdx.x;
  for (int l = 0; l < Lc; ++l) {
    const float* wi = w_ih + (size_t)l * 4*Hc * Dc;
    const float* wh = w_hh + (size_t)l * 4*Hc * Hc;
    const float* bi = b_ih + l * 4*Hc;
    const float* bh = b_hh + l * 4*Hc;
    if (tid < Hc) { hv[tid] = 0.0f; cv[tid] = 0.0f; }
    __syncthreads();
    for (int u = 0; u < Uc; ++u) {
      if (tid < Dc) {
        xv[tid] = (l == 0) ? emb[(size_t)ids[b*Uc + u] * Dc + tid]
                           : ys[(size_t)(b*Uc + u) * Hc + tid];
      }
      __syncthreads();
      // each thread computes 2 gates (2048 gates / 1024 threads)
      const int g0 = tid * 2;
      float acc0 = bi[g0]     + bh[g0];
      float acc1 = bi[g0 + 1] + bh[g0 + 1];
      const float4* wi0 = (const float4*)(wi + (size_t)g0 * Dc);
      const float4* wi1 = (const float4*)(wi + (size_t)(g0 + 1) * Dc);
      const float4* wh0 = (const float4*)(wh + (size_t)g0 * Hc);
      const float4* wh1 = (const float4*)(wh + (size_t)(g0 + 1) * Hc);
      #pragma unroll 4
      for (int k4 = 0; k4 < Dc/4; ++k4) {
        float4 a0 = wi0[k4], a1 = wi1[k4], c0 = wh0[k4], c1 = wh1[k4];
        float x0 = xv[4*k4+0], x1 = xv[4*k4+1], x2 = xv[4*k4+2], x3 = xv[4*k4+3];
        float h0 = hv[4*k4+0], h1 = hv[4*k4+1], h2 = hv[4*k4+2], h3 = hv[4*k4+3];
        acc0 += a0.x*x0 + a0.y*x1 + a0.z*x2 + a0.w*x3
              + c0.x*h0 + c0.y*h1 + c0.z*h2 + c0.w*h3;
        acc1 += a1.x*x0 + a1.y*x1 + a1.z*x2 + a1.w*x3
              + c1.x*h0 + c1.y*h1 + c1.z*h2 + c1.w*h3;
      }
      gates[g0] = acc0; gates[g0 + 1] = acc1;
      __syncthreads();
      if (tid < Hc) {
        float ig = sigf(gates[tid]);
        float fg = sigf(gates[Hc + tid]);
        float gg = tanhf(gates[2*Hc + tid]);
        float og = sigf(gates[3*Hc + tid]);
        float c = fg * cv[tid] + ig * gg;
        float h = og * tanhf(c);
        cv[tid] = c; hv[tid] = h;
        ys[(size_t)(b*Uc + u) * Hc + tid] = h;   // layer output (in-place over input seq)
      }
      __syncthreads();
    }
    if (tid < Hc) {
      hc_out[l * (Bc*Hc) + b * Hc + tid] = hv[tid];                 // h (L,B,H)
      hc_out[Lc*Bc*Hc + l * (Bc*Hc) + b * Hc + tid] = cv[tid];      // c (L,B,H)
    }
    __syncthreads();
  }
}

// ---- kernel 2: fc projection + dec_p (= x@fc_w.T+fc_b then @wd.T + jb1) ----
__global__ __launch_bounds__(512)
void k_fcdec(const float* __restrict__ ys, const float* __restrict__ fc_w,
             const float* __restrict__ fc_b, const float* __restrict__ jw1,
             const float* __restrict__ jb1, float* __restrict__ dec_p) {
  __shared__ float xr[Hc], xf[Dc];
  const int bu = blockIdx.x, tid = threadIdx.x;
  xr[tid] = ys[(size_t)bu * Hc + tid];
  __syncthreads();
  float acc = fc_b[tid];
  const float4* wr = (const float4*)(fc_w + (size_t)tid * Hc);
  #pragma unroll 4
  for (int k4 = 0; k4 < Hc/4; ++k4) {
    float4 w4 = wr[k4];
    acc += w4.x*xr[4*k4] + w4.y*xr[4*k4+1] + w4.z*xr[4*k4+2] + w4.w*xr[4*k4+3];
  }
  xf[tid] = acc;
  __syncthreads();
  float acc2 = jb1[tid];   // fold jb1 into dec_p
  const float4* wd = (const float4*)(jw1 + (size_t)tid * (2*Dc) + Dc);
  #pragma unroll 4
  for (int k4 = 0; k4 < Dc/4; ++k4) {
    float4 w4 = wd[k4];
    acc2 += w4.x*xf[4*k4] + w4.y*xf[4*k4+1] + w4.z*xf[4*k4+2] + w4.w*xf[4*k4+3];
  }
  dec_p[(size_t)bu * Hc + tid] = acc2;
}

// ---- kernel 3: enc_p = memory @ we.T, WMMA f16 (M=2000, N=512, K=512) ------
__global__ __launch_bounds__(256)
void k_encp(const float* __restrict__ memory, const float* __restrict__ jw1,
            float* __restrict__ enc_p) {
  const int wave = threadIdx.x >> 5, lane = threadIdx.x & 31;
  const int wid = blockIdx.x * 8 + wave;       // 0..3999 (exact)
  const int mtile = wid >> 5;                  // 125 M-tiles
  const int ntile = wid & 31;                  // 32 N-tiles
  const int halfsel = lane >> 4, lrow = lane & 15, kb = halfsel * 8;
  const float* arow = memory + (size_t)(mtile*16 + lrow) * Dc;
  const float* brow = jw1 + (size_t)(ntile*16 + lrow) * (2*Dc);  // we = jw1[:, :D]
  v8f acc = {};
  for (int k0 = 0; k0 < Dc; k0 += 32) {
    __builtin_prefetch(arow + k0 + 64, 0, 1);   // global_prefetch_b8
    v16h a = load_frag_f32(arow + k0, kb);
    v16h bfr = load_frag_f32(brow + k0, kb);
    acc = __builtin_amdgcn_wmma_f32_16x16x32_f16(false, a, false, bfr,
                                                 (short)0, acc, false, false);
  }
  const int col = ntile*16 + lrow;
  const int mbase = mtile*16 + halfsel*8;
  #pragma unroll
  for (int i = 0; i < 8; ++i)
    enc_p[(size_t)(mbase + i) * Hc + col] = acc[i];
}

// ---- kernel 4: fused joint: out = tanh(enc+dec) @ jw2.T + jb2 (WMMA f16) ---
__global__ __launch_bounds__(256)
void k_joint(const float* __restrict__ enc_p, const float* __restrict__ dec_p,
             const float* __restrict__ jw2, const float* __restrict__ jb2,
             float* __restrict__ out) {
  __shared__ __align__(16) _Float16 w2s[Vc * Hc];   // 64KB: jw2 as f16, LDS-resident
  const int tid = threadIdx.x;
  for (int i = tid; i < Vc * Hc; i += 256) w2s[i] = (_Float16)jw2[i];
  __syncthreads();

  const int wave = tid >> 5, lane = tid & 31;
  const int tile = blockIdx.x * 8 + wave;
  if (tile >= MTILES) return;                  // wave-uniform: EXEC all-ones below
  const int halfsel = lane >> 4, lrow = lane & 15, kb = halfsel * 8;

  const int gr = tile * 16 + lrow;             // global (b,t,u) row for A fragment
  const int b = gr / TU, rem = gr % TU, t = rem / Uc, u = rem % Uc;
  const float* er = enc_p + (size_t)(b*Tc + t) * Hc;
  const float* dr = dec_p + (size_t)(b*Uc + u) * Hc;

  v8f zero = {};
  v8f acc[4];
  #pragma unroll
  for (int nt = 0; nt < 4; ++nt) acc[nt] = zero;

  for (int k0 = 0; k0 < Hc; k0 += 32) {
    __builtin_prefetch(er + k0 + 128, 0, 1);    // stream next enc/dec chunks
    __builtin_prefetch(dr + k0 + 128, 0, 1);
    const float4* eq = (const float4*)(er + k0);
    const float4* dq = (const float4*)(dr + k0);
    const int i0 = kb >> 2;
    float4 e0 = eq[i0], e1 = eq[i0+1], e2 = eq[i0+4], e3 = eq[i0+5];
    float4 d0 = dq[i0], d1 = dq[i0+1], d2 = dq[i0+4], d3 = dq[i0+5];
    v16h a;
    a[0]=(_Float16)fast_tanh(e0.x+d0.x); a[1]=(_Float16)fast_tanh(e0.y+d0.y);
    a[2]=(_Float16)fast_tanh(e0.z+d0.z); a[3]=(_Float16)fast_tanh(e0.w+d0.w);
    a[4]=(_Float16)fast_tanh(e1.x+d1.x); a[5]=(_Float16)fast_tanh(e1.y+d1.y);
    a[6]=(_Float16)fast_tanh(e1.z+d1.z); a[7]=(_Float16)fast_tanh(e1.w+d1.w);
    a[8]=(_Float16)fast_tanh(e2.x+d2.x); a[9]=(_Float16)fast_tanh(e2.y+d2.y);
    a[10]=(_Float16)fast_tanh(e2.z+d2.z); a[11]=(_Float16)fast_tanh(e2.w+d2.w);
    a[12]=(_Float16)fast_tanh(e3.x+d3.x); a[13]=(_Float16)fast_tanh(e3.y+d3.y);
    a[14]=(_Float16)fast_tanh(e3.z+d3.z); a[15]=(_Float16)fast_tanh(e3.w+d3.w);
    #pragma unroll
    for (int nt = 0; nt < 4; ++nt) {
      const _Float16* wrow = w2s + (size_t)(nt*16 + lrow) * Hc + k0 + kb;
      v8h lo = *(const v8h*)wrow;          // k = kb..kb+7
      v8h hi = *(const v8h*)(wrow + 16);   // k = kb+16..kb+23
      v16h bfr = __builtin_shufflevector(lo, hi, 0,1,2,3,4,5,6,7,8,9,10,11,12,13,14,15);
      acc[nt] = __builtin_amdgcn_wmma_f32_16x16x32_f16(false, a, false, bfr,
                                                       (short)0, acc[nt], false, false);
    }
  }
  #pragma unroll
  for (int nt = 0; nt < 4; ++nt) {
    const int col = nt*16 + lrow;
    const float bias = jb2[col];
    #pragma unroll
    for (int i = 0; i < 8; ++i) {
      const int mr = tile*16 + halfsel*8 + i;
      const int b2 = mr / TU, r2 = mr % TU, t2 = r2 / Uc, u2 = r2 % Uc;
      out[(size_t)((b2*Tc + t2)*Uc + u2) * Vc + col] = acc[nt][i] + bias;
    }
  }
}

// ---- launcher --------------------------------------------------------------
extern "C" void kernel_launch(void* const* d_in, const int* in_sizes, int n_in,
                              void* d_out, int out_size, void* d_ws, size_t ws_size,
                              hipStream_t stream) {
  const int*   ids   = (const int*)  d_in[0];
  const float* mem   = (const float*)d_in[1];
  const float* emb   = (const float*)d_in[2];
  const float* w_ih  = (const float*)d_in[3];
  const float* w_hh  = (const float*)d_in[4];
  const float* b_ih  = (const float*)d_in[5];
  const float* b_hh  = (const float*)d_in[6];
  const float* fc_w  = (const float*)d_in[7];
  const float* fc_b  = (const float*)d_in[8];
  const float* jw1   = (const float*)d_in[9];
  const float* jb1   = (const float*)d_in[10];
  const float* jw2   = (const float*)d_in[11];
  const float* jb2   = (const float*)d_in[12];
  float* out = (float*)d_out;

  // workspace: enc_p (B*T*H) | dec_p (B*U*H) | ys (B*U*H)
  float* enc_p = (float*)d_ws;
  float* dec_p = enc_p + (size_t)Bc*Tc*Hc;       // +1,024,000
  float* ys    = dec_p + (size_t)Bc*Uc*Hc;       // +204,800

  k_lstm <<<Bc, 1024, 0, stream>>>(ids, emb, w_ih, w_hh, b_ih, b_hh, ys, out + OUT0);
  k_fcdec<<<Bc*Uc, 512, 0, stream>>>(ys, fc_w, fc_b, jw1, jb1, dec_p);
  k_encp <<<(Bc*Tc/16)*(Hc/16)/8, 256, 0, stream>>>(mem, jw1, enc_p);
  k_joint<<<(MTILES + 7)/8, 256, 0, stream>>>(enc_p, dec_p, jw2, jb2, out);
}
